// RuleGraphConvLayer_49357764165669
// MI455X (gfx1250) — matrix operand order repaired and checked by reading, not compiled, not run on hardware
//
#include <hip/hip_runtime.h>
#include <hip/hip_bf16.h>

// CDNA5 / gfx1250: wave32, WMMA 16x16x32 bf16 (f32 accumulate), async global->LDS copies.
typedef __bf16 bf16_t;
typedef __attribute__((ext_vector_type(16))) __bf16 v16bf;
typedef __attribute__((ext_vector_type(8)))  float  v8f;

#define NF  81            // node feature dim
#define NB  22            // bond dim
#define KEDGE (NF + NB)   // 103
#define OC  64            // output channels
#define KE_TILES 4        // edge GEMM: K padded 103 -> 128 (4 x 32)
#define KN_TILES 3        // node GEMM: K padded 81  -> 96  (3 x 32)
#define LDA_E 136         // LDS row stride (bf16) for 128-wide tile (+8 = 16B shift/row, avoids bank conflicts)
#define LDA_N 104         // LDS row stride for 96-wide tile
#define EPB 32            // edges per block (2 M-subtiles of 16)

union V16 {
    v16bf v;
    uint4 u[2];
};

// A-fragment load, 16x32 tile in contiguous k order (same order as the B packing):
//   lanes 0-15 : row M=lane,    elems 0..15 = K kb+0..15
//   lanes 16-31: row M=lane-16, elems 0..15 = K kb+16..31
__device__ inline v16bf load_a_frag(const bf16_t* lds, int lda, int kb, int lane) {
    const int row  = lane & 15;
    const int half = lane >> 4;
    V16 r;
    r.u[0] = *(const uint4*)(lds + row * lda + kb + half * 16);
    r.u[1] = *(const uint4*)(lds + row * lda + kb + half * 16 + 8);
    return r.v;
}

// B fragments pre-packed in global scratch as [ktile][ntile][lane][16] bf16:
// lane holds column n = nt*16 + (lane&15), K = kt*32 + (lane>=16 ? 16 : 0) + i.
__device__ inline v16bf load_b_frag(const bf16_t* p) {
    V16 r;
    r.u[0] = *(const uint4*)(p);
    r.u[1] = *(const uint4*)(p + 8);
    return r.v;
}

// ---------------- prep: repack w_n (103x64) and w_s (81x64) into B-frag layout ----------------
__global__ void prep_weights(const float* __restrict__ w_n, const float* __restrict__ w_s,
                             bf16_t* __restrict__ ws_b, bf16_t* __restrict__ ws_s) {
    const int t      = blockIdx.x * blockDim.x + threadIdx.x;
    const int stride = gridDim.x * blockDim.x;
    const int totalB = KE_TILES * 4 * 32 * 16;
    for (int idx = t; idx < totalB; idx += stride) {
        int i    = idx & 15;
        int lane = (idx >> 4) & 31;
        int nt   = (idx >> 9) & 3;
        int kt   = idx >> 11;
        int n = nt * 16 + (lane & 15);
        int k = kt * 32 + ((lane >> 4) ? 16 : 0) + i;
        float v = (k < KEDGE) ? w_n[k * OC + n] : 0.0f;
        ws_b[idx] = (bf16_t)v;
    }
    const int totalS = KN_TILES * 4 * 32 * 16;
    for (int idx = t; idx < totalS; idx += stride) {
        int i    = idx & 15;
        int lane = (idx >> 4) & 31;
        int nt   = (idx >> 9) & 3;
        int kt   = idx >> 11;
        int n = nt * 16 + (lane & 15);
        int k = kt * 32 + ((lane >> 4) ? 16 : 0) + i;
        float v = (k < NF) ? w_s[k * OC + n] : 0.0f;
        ws_s[idx] = (bf16_t)v;
    }
}

// ---------------- node GEMM: out = features @ w_s (initializes d_out) ----------------
// 128 threads = 4 waves; block handles 16 nodes x 64 cols; wave w owns n-tile w.
__global__ void __launch_bounds__(128)
node_gemm(const float* __restrict__ features, const bf16_t* __restrict__ ws_s,
          float* __restrict__ out, int n_nodes) {
    __shared__ bf16_t ldsA[16 * LDA_N];
    const int tid  = threadIdx.x;
    const int lane = tid & 31;
    const int w    = tid >> 5;
    const int base = blockIdx.x * 16;
    const bool full = (base + 16 <= n_nodes);

    // Stage A tile: 16 rows x 96 cols bf16 (cols 81..95 zero). 8 k-groups x 12 cols.
    {
        int row  = tid & 15;
        int kg   = tid >> 4;
        int node = base + row;
        if (node >= n_nodes) node = n_nodes - 1;
        const float* fr = features + (long)node * NF;
        #pragma unroll
        for (int i = 0; i < 12; ++i) {
            int k = kg * 12 + i;
            float v = (k < NF) ? fr[k] : 0.0f;
            ldsA[row * LDA_N + k] = (bf16_t)v;
        }
    }
    __syncthreads();

    v8f c = {};
    #pragma unroll
    for (int kt = 0; kt < KN_TILES; ++kt) {
        v16bf a = load_a_frag(ldsA, LDA_N, kt * 32, lane);
        v16bf b = load_b_frag(ws_s + ((kt * 4 + w) * 32 + lane) * 16);
        c = __builtin_amdgcn_wmma_f32_16x16x32_bf16(false, a, false, b, (short)0, c, false, false);
    }

    // C/D layout: vgpr j -> row j + 8*half; lane -> col (lane&15) within n-tile.
    const int col   = w * 16 + (lane & 15);
    const int rbase = base + ((lane >> 4) ? 8 : 0);
    if (full) {
        #pragma unroll
        for (int j = 0; j < 8; ++j)
            out[(long)(rbase + j) * OC + col] = c[j];
    } else {
        #pragma unroll
        for (int j = 0; j < 8; ++j) {
            int r = rbase + j;
            if (r < n_nodes) out[(long)r * OC + col] = c[j];
        }
    }
}

// ---------------- edge GEMM + segment-sum: out[src[e]] += edge_vec(e) @ w_n ----------------
// 256 threads = 8 waves; block handles 32 edges (2 M-subtiles) x 64 cols.
// Bond slab (contiguous 32x22 f32 = 2816B) is copied via async global->LDS b128,
// overlapped with the random feature gathers, drained by s_wait_asynccnt.
__global__ void __launch_bounds__(256)
edge_gemm(const float* __restrict__ features, const float* __restrict__ bond,
          const int* __restrict__ src, const int* __restrict__ dst,
          const bf16_t* __restrict__ ws_b, float* __restrict__ out, int n_edges) {
    __shared__ bf16_t ldsA[EPB * LDA_E];
    __shared__ float  ldsBond[EPB * NB];
    __shared__ int    ldsSrc[EPB];
    const int  tid  = threadIdx.x;
    const int  lane = tid & 31;
    const int  w    = tid >> 5;               // 8 waves
    const long base = (long)blockIdx.x * EPB;
    const bool full = (base + EPB <= n_edges);

    // Kick off async copy of the bond slab: 2816B = 176 x b128, one chunk per lane.
    if (full && tid < (EPB * NB * 4) / 16) {
        uint32_t la = (uint32_t)(uintptr_t)(&ldsBond[0]) + tid * 16;
        uint64_t ga = (uint64_t)(uintptr_t)(bond + base * NB) + (uint64_t)tid * 16;
        asm volatile("global_load_async_to_lds_b128 %0, %1, off"
                     :: "v"(la), "v"(ga) : "memory");
    }

    // Phase 1: feature part of the edge-vector tile (k<81) + zero pad (k>=103).
    {
        int  el = tid & (EPB - 1);
        int  kg = tid >> 5;                   // 8 k-groups x 16 K values
        long e  = base + el;
        long ec = (e < n_edges) ? e : (long)(n_edges - 1);
        int  s  = src[ec];
        int  dn = dst[ec];
        if (kg == 0) ldsSrc[el] = s;          // kg==0 threads cover el = tid
        const float* sf = features + (long)s  * NF;
        const float* nf = features + (long)dn * NF;
        float dx = sf[0] - nf[0], dy = sf[1] - nf[1], dz = sf[2] - nf[2];
        float d2 = dx * dx + dy * dy + dz * dz;
        float inv = (d2 > 0.0f) ? (1.0f / d2) : 10000.0f;  // dist==0 -> d=0.01 -> 1/d^2 = 1e4
        #pragma unroll
        for (int i = 0; i < 16; ++i) {
            int k = kg * 16 + i;
            if (k < 3)           ldsA[el * LDA_E + k] = (bf16_t)(nf[k] * inv);
            else if (k < NF)     ldsA[el * LDA_E + k] = (bf16_t)((sf[k] + nf[k]) * inv);
            else if (k >= KEDGE) ldsA[el * LDA_E + k] = (bf16_t)0.0f;
            // k in [81,103) filled in phase 2 from the async bond slab
        }
    }
    asm volatile("s_wait_asynccnt 0x0" ::: "memory");
    __syncthreads();

    // Phase 2: repack bond slab (f32 in LDS) into bf16 columns 81..102 of the A tile.
    for (int idx = tid; idx < EPB * NB; idx += 256) {
        int el2 = idx / NB;
        int j   = idx - el2 * NB;
        float v;
        if (full) {
            v = ldsBond[idx];
        } else {
            long ee = base + el2;
            long ec2 = (ee < n_edges) ? ee : (long)(n_edges - 1);
            v = bond[ec2 * NB + j];
        }
        ldsA[el2 * LDA_E + NF + j] = (bf16_t)v;
    }
    __syncthreads();

    // WMMA phase: wave w -> edge-subtile st = w>>2, n-tile nt = w&3.
    const int st = w >> 2;
    const int nt = w & 3;
    const bf16_t* ldsAs = ldsA + st * 16 * LDA_E;
    v8f c = {};
    #pragma unroll
    for (int kt = 0; kt < KE_TILES; ++kt) {
        v16bf a = load_a_frag(ldsAs, LDA_E, kt * 32, lane);
        v16bf b = load_b_frag(ws_b + ((kt * 4 + nt) * 32 + lane) * 16);
        c = __builtin_amdgcn_wmma_f32_16x16x32_bf16(false, a, false, b, (short)0, c, false, false);
    }

    // Scatter-accumulate: out[src[edge]][col] += c[j]  (global_atomic_add_f32, no return)
    const int col  = nt * 16 + (lane & 15);
    const int mofs = st * 16 + ((lane >> 4) ? 8 : 0);
    if (full) {
        #pragma unroll
        for (int j = 0; j < 8; ++j)
            atomicAdd(out + (long)ldsSrc[mofs + j] * OC + col, c[j]);
    } else {
        #pragma unroll
        for (int j = 0; j < 8; ++j) {
            long ee = base + mofs + j;
            if (ee < n_edges)
                atomicAdd(out + (long)ldsSrc[mofs + j] * OC + col, c[j]);
        }
    }
}

extern "C" void kernel_launch(void* const* d_in, const int* in_sizes, int n_in,
                              void* d_out, int out_size, void* d_ws, size_t ws_size,
                              hipStream_t stream) {
    const float* features = (const float*)d_in[0];
    const float* bond     = (const float*)d_in[1];
    const float* w_s      = (const float*)d_in[2];
    const float* w_n      = (const float*)d_in[3];
    const int*   src      = (const int*)d_in[4];
    const int*   dst      = (const int*)d_in[5];
    float*       out      = (float*)d_out;

    const int n_nodes = in_sizes[0] / NF;
    const int n_edges = in_sizes[4];

    bf16_t* ws_b = (bf16_t*)d_ws;                      // 4*4*32*16 bf16 = 16 KB
    bf16_t* ws_s = ws_b + KE_TILES * 4 * 32 * 16;      // 3*4*32*16 bf16 = 12 KB

    prep_weights<<<dim3(16), dim3(256), 0, stream>>>(w_n, w_s, ws_b, ws_s);

    const int nblocks = (n_nodes + 15) / 16;
    node_gemm<<<dim3(nblocks), dim3(128), 0, stream>>>(features, ws_s, out, n_nodes);

    const int eblocks = (n_edges + EPB - 1) / EPB;
    edge_gemm<<<dim3(eblocks), dim3(256), 0, stream>>>(features, bond, src, dst, ws_b, out, n_edges);
}
